// MultiHeadAttention_54176717471843
// MI455X (gfx1250) — matrix operand order repaired and checked by reading, not compiled, not run on hardware
//
#include <hip/hip_runtime.h>

// ---------------------------------------------------------------------------
// MI455X (gfx1250) fused multi-head attention, bf16 WMMA pipeline.
//   1) fp32 -> bf16 conversion of activations + weights (one pass)
//   2) WMMA GEMM projections with async global->LDS double-buffered staging
//      (K, Q row-major bf16; V stored transposed per head)
//   3) flash-attention fused QK^T -> online softmax -> P*V (all WMMA)
//   4) WMMA GEMM output projection -> fp32 d_out
// ---------------------------------------------------------------------------

typedef __attribute__((ext_vector_type(16))) __bf16 v16bf;
typedef __attribute__((ext_vector_type(8)))  __bf16 v8bf;
typedef __attribute__((ext_vector_type(4)))  __bf16 v4bf;
typedef __attribute__((ext_vector_type(8)))  float  v8f;

#define B_   4
#define T_   2048
#define DM_  256
#define DK_  32
#define H_   8
#define BT_  (B_ * T_)       // 8192
#define HDM_ (H_ * DM_)      // 2048 (v-projection channels, also emb channels)
#define HDK_ (H_ * DK_)      // 256  (k/q-projection channels)

#define LDSR 48              // padded LDS row stride in bf16 elems (96B)

__device__ __forceinline__ v8f v8f_zero() {
  v8f z;
#pragma unroll
  for (int i = 0; i < 8; ++i) z[i] = 0.0f;
  return z;
}

// 16-bit A-fragment (16x32): per lane, elements are K = [h*16 .. h*16+15]
// (contiguous, matching the contiguous B loads). rowp points at (row, k0).
__device__ __forceinline__ v16bf load_a_frag(const __bf16* rowp, int h16) {
  v8bf lo = *(const v8bf*)(rowp + h16 * 16);
  v8bf hi = *(const v8bf*)(rowp + h16 * 16 + 8);
  return __builtin_shufflevector(lo, hi, 0, 1, 2, 3, 4, 5, 6, 7,
                                 8, 9, 10, 11, 12, 13, 14, 15);
}

__device__ __forceinline__ v8f wmma_bf16(v16bf a, v16bf b, v8f c) {
  return __builtin_amdgcn_wmma_f32_16x16x32_bf16(false, a, false, b,
                                                 (short)0, c, false, false);
}

// Async DMA: 16 bytes global -> LDS, tracked by ASYNCcnt (bypasses VGPRs).
// lds_off = workgroup-relative LDS byte offset (low 32 bits of generic ptr:
// the LDS aperture occupies ADDR[63:32], ADDR[31:0] is the LDS offset).
__device__ __forceinline__ void async_copy16(unsigned int lds_off,
                                             const __bf16* gptr) {
  asm volatile("global_load_async_to_lds_b128 %0, %1, off"
               :: "v"(lds_off), "v"(gptr)
               : "memory");
}

__device__ __forceinline__ void wait_async0() {
  asm volatile("s_wait_asynccnt 0x0" ::: "memory");
}

// ---------------------------------------------------------------------------
// fp32 -> bf16 conversion (vectorized x4)
// ---------------------------------------------------------------------------
__global__ __launch_bounds__(256) void cvt_f32_bf16(const float* __restrict__ src,
                                                    __bf16* __restrict__ dst,
                                                    int n4) {
  int i = blockIdx.x * blockDim.x + threadIdx.x;
  if (i >= n4) return;
  float4 v = ((const float4*)src)[i];
  v4bf o;
  o[0] = (__bf16)v.x; o[1] = (__bf16)v.y; o[2] = (__bf16)v.z; o[3] = (__bf16)v.w;
  ((v4bf*)dst)[i] = o;
}

// ---------------------------------------------------------------------------
// Tiled WMMA GEMM: out[M,N] = A[M,K] * W[N,K]^T + bias[N]
// Block = 256 threads (8 waves), tile 128x128.
// K staged in 32-chunks through double-buffered LDS filled by async DMA:
// stage k+1's copies are issued right after the barrier, overlapping the
// 8-wmma compute of stage k; one barrier per K-step.
// Wave (2x4 grid) computes a 64x32 sub-tile = 4x2 wmma accumulators.
// MODE: 0 = bf16 row-major, 1 = fp32 row-major, 2 = V-transpose store
//       (out[b][h][d][t] bf16, with n = h*256+d, m = b*2048+t)
// ---------------------------------------------------------------------------
template <int MODE>
__global__ __launch_bounds__(256) void gemm_bias_kernel(
    const __bf16* __restrict__ A, const __bf16* __restrict__ W,
    const float* __restrict__ bias, void* __restrict__ outp,
    int N, int K) {
  __shared__ __bf16 As[2][128 * LDSR];
  __shared__ __bf16 Bs[2][128 * LDSR];

  const int tid  = threadIdx.x;
  const int lane = tid & 31;
  const int wave = tid >> 5;
  const int wm   = wave >> 2;   // 0..1  (64 rows)
  const int wn   = wave & 3;    // 0..3  (32 cols)
  const int r16  = lane & 15;
  const int h16  = lane >> 4;

  const int m0 = blockIdx.x * 128;
  const int n0 = blockIdx.y * 128;

  v8f acc[4][2];
#pragma unroll
  for (int mt = 0; mt < 4; ++mt)
#pragma unroll
    for (int nt = 0; nt < 2; ++nt) acc[mt][nt] = v8f_zero();

  // staging map: thread -> (row 0..127, 16-bf16 half)
  const int lrow  = tid >> 1;
  const int lhalf = tid & 1;
  const __bf16* ga = A + (size_t)(m0 + lrow) * K + lhalf * 16;
  const __bf16* gb = W + (size_t)(n0 + lrow) * K + lhalf * 16;
  const unsigned int la =
      (unsigned int)(uintptr_t)&As[0][lrow * LDSR + lhalf * 16];
  const unsigned int lb =
      (unsigned int)(uintptr_t)&Bs[0][lrow * LDSR + lhalf * 16];
  const unsigned int bufstride = 128 * LDSR * sizeof(__bf16);  // 12288 B

  // prefetch stage 0 into buffer 0 (2 x 16B per matrix per thread)
  async_copy16(la, ga);
  async_copy16(la + 16, ga + 8);
  async_copy16(lb, gb);
  async_copy16(lb + 16, gb + 8);

  const int nk = K / 32;
  for (int ki = 0; ki < nk; ++ki) {
    const int cur = ki & 1;
    wait_async0();       // my 4 async copies of stage ki have landed
    __syncthreads();     // everyone's landed; everyone done reading buf[cur^1]

    if (ki + 1 < nk) {   // prefetch stage ki+1, overlapped with compute below
      const unsigned int nb = (unsigned int)(cur ^ 1) * bufstride;
      const __bf16* ga2 = ga + (size_t)(ki + 1) * 32;
      const __bf16* gb2 = gb + (size_t)(ki + 1) * 32;
      async_copy16(la + nb, ga2);
      async_copy16(la + nb + 16, ga2 + 8);
      async_copy16(lb + nb, gb2);
      async_copy16(lb + nb + 16, gb2 + 8);
    }

    const __bf16* asrc = &As[cur][0];
    const __bf16* bsrc = &Bs[cur][0];
    v16bf af[4], bf2[2];
#pragma unroll
    for (int mt = 0; mt < 4; ++mt)
      af[mt] = load_a_frag(&asrc[(wm * 64 + mt * 16 + r16) * LDSR], h16);
#pragma unroll
    for (int nt = 0; nt < 2; ++nt)
      bf2[nt] = *(const v16bf*)&bsrc[(wn * 32 + nt * 16 + r16) * LDSR + h16 * 16];
#pragma unroll
    for (int mt = 0; mt < 4; ++mt)
#pragma unroll
      for (int nt = 0; nt < 2; ++nt)
        acc[mt][nt] = wmma_bf16(af[mt], bf2[nt], acc[mt][nt]);
  }

  // epilogue: bias + store (C-layout: VGPR j -> row j + 8*h16, col = lane&15)
#pragma unroll
  for (int nt = 0; nt < 2; ++nt) {
    const int n  = n0 + wn * 32 + nt * 16 + r16;
    const float bv = bias[n];
#pragma unroll
    for (int mt = 0; mt < 4; ++mt) {
#pragma unroll
      for (int j = 0; j < 8; ++j) {
        const int m  = m0 + wm * 64 + mt * 16 + j + 8 * h16;
        const float val = acc[mt][nt][j] + bv;
        if (MODE == 0) {
          ((__bf16*)outp)[(size_t)m * N + n] = (__bf16)val;
        } else if (MODE == 1) {
          ((float*)outp)[(size_t)m * N + n] = val;
        } else {  // V-transpose: Vt[b][h][d][t]
          const int bb = m >> 11, t = m & (T_ - 1);
          const int hh = n >> 8,  d = n & (DM_ - 1);
          ((__bf16*)outp)[(((size_t)(bb * H_ + hh) * DM_ + d) * T_) + t] = (__bf16)val;
        }
      }
    }
  }
}

// ---------------------------------------------------------------------------
// Fused flash attention, one (b, h, 128-query) tile per block; 8 waves,
// each wave owns 16 queries. Keys streamed in blocks of 32 (= one WMMA K).
//   S[q,s] = Q[q,:] . K[s,:]           (dk = 32, unscaled, per reference)
//   online softmax over s, P*V accumulated in fp32 (256 v-dims = 16 wmmas)
// Output stored as Emb[b][q][d*8 + h] (reference's bqdh flatten).
// ---------------------------------------------------------------------------
__global__ __launch_bounds__(256) void flash_attn_kernel(
    const __bf16* __restrict__ Qp, const __bf16* __restrict__ Kp,
    const __bf16* __restrict__ Vt, __bf16* __restrict__ Emb) {
  __shared__ __bf16 Ps[8 * 16 * LDSR];  // per-wave P transpose scratch

  const int tid  = threadIdx.x;
  const int lane = tid & 31;
  const int wave = tid >> 5;
  const int r16  = lane & 15;
  const int h16  = lane >> 4;

  const int bb = blockIdx.z;
  const int hh = blockIdx.y;
  const int q0 = blockIdx.x * 128 + wave * 16;

  // Q A-fragment: row q0+r16, k = hh*32 + (frag pattern)
  const v16bf qa =
      load_a_frag(Qp + ((size_t)(bb * T_ + q0 + r16)) * HDK_ + hh * DK_, h16);

  const __bf16* kbase = Kp + (size_t)bb * T_ * HDK_ + hh * DK_;
  const __bf16* vbase = Vt + (size_t)(bb * H_ + hh) * DM_ * T_;
  __bf16* pl = &Ps[wave * 16 * LDSR];

  v8f o[16];
#pragma unroll
  for (int nt = 0; nt < 16; ++nt) o[nt] = v8f_zero();
  float mrow[8], lrow[8];
#pragma unroll
  for (int j = 0; j < 8; ++j) { mrow[j] = -3.0e38f; lrow[j] = 0.0f; }

  for (int s0 = 0; s0 < T_; s0 += 32) {
    // --- S = Q * K^T : two 16x16 tiles (s-local 0..15, 16..31) ---
    const v16bf kb0 =
        *(const v16bf*)(kbase + (size_t)(s0 + r16) * HDK_ + h16 * 16);
    const v16bf kb1 =
        *(const v16bf*)(kbase + (size_t)(s0 + 16 + r16) * HDK_ + h16 * 16);
    v8f s0a = wmma_bf16(qa, kb0, v8f_zero());
    v8f s1a = wmma_bf16(qa, kb1, v8f_zero());

    // --- row max over 32 keys (16-lane half-wave reduction) ---
    float mnew[8];
#pragma unroll
    for (int j = 0; j < 8; ++j) mnew[j] = fmaxf(s0a[j], s1a[j]);
#pragma unroll
    for (int off = 1; off < 16; off <<= 1)
#pragma unroll
      for (int j = 0; j < 8; ++j)
        mnew[j] = fmaxf(mnew[j], __shfl_xor(mnew[j], off, 32));

    // --- rescale running state ---
    float fac[8];
#pragma unroll
    for (int j = 0; j < 8; ++j) {
      const float mn = fmaxf(mrow[j], mnew[j]);
      fac[j]  = __expf(mrow[j] - mn);
      mrow[j] = mn;
      lrow[j] *= fac[j];
    }
#pragma unroll
    for (int nt = 0; nt < 16; ++nt)
#pragma unroll
      for (int j = 0; j < 8; ++j) o[nt][j] *= fac[j];

    // --- P = exp(S - m); row-sum; transpose into A-fragment via LDS ---
    float rs[8];
#pragma unroll
    for (int j = 0; j < 8; ++j) {
      const float p0 = __expf(s0a[j] - mrow[j]);
      const float p1 = __expf(s1a[j] - mrow[j]);
      rs[j] = p0 + p1;
      const int prow = j + 8 * h16;
      pl[prow * LDSR + r16]      = (__bf16)p0;
      pl[prow * LDSR + 16 + r16] = (__bf16)p1;
    }
#pragma unroll
    for (int off = 1; off < 16; off <<= 1)
#pragma unroll
      for (int j = 0; j < 8; ++j) rs[j] += __shfl_xor(rs[j], off, 32);
#pragma unroll
    for (int j = 0; j < 8; ++j) lrow[j] += rs[j];

    // DS ops are in-order within a wave: transpose read is safe w/o barrier.
    const v16bf pa = load_a_frag(&pl[r16 * LDSR], h16);

    // --- O += P * V  (V^T rows give contiguous s-runs per d-lane) ---
#pragma unroll
    for (int nt = 0; nt < 16; ++nt) {
      const v16bf vb =
          *(const v16bf*)(vbase + (size_t)(nt * 16 + r16) * T_ + s0 + h16 * 16);
      o[nt] = wmma_bf16(pa, vb, o[nt]);
    }
  }

  // --- epilogue: normalize, store Emb[b][q][d*8 + h] ---
#pragma unroll
  for (int j = 0; j < 8; ++j) {
    const int q = q0 + j + 8 * h16;
    const float inv = 1.0f / lrow[j];
    __bf16* erow = Emb + ((size_t)(bb * T_ + q)) * HDM_ + hh;
#pragma unroll
    for (int nt = 0; nt < 16; ++nt) {
      const int d = nt * 16 + r16;
      erow[(size_t)d * H_] = (__bf16)(o[nt][j] * inv);
    }
  }
}

// ---------------------------------------------------------------------------
extern "C" void kernel_launch(void* const* d_in, const int* in_sizes, int n_in,
                              void* d_out, int out_size, void* d_ws, size_t ws_size,
                              hipStream_t stream) {
  const float* key_in = (const float*)d_in[0];
  const float* qry_in = (const float*)d_in[1];
  const float* val_in = (const float*)d_in[2];
  // d_in[3] = attention_mask (unused by reference)
  const float* WKw = (const float*)d_in[4];
  const float* WKb = (const float*)d_in[5];
  const float* WQw = (const float*)d_in[6];
  const float* WQb = (const float*)d_in[7];
  const float* WVw = (const float*)d_in[8];
  const float* WVb = (const float*)d_in[9];
  const float* WOw = (const float*)d_in[10];
  const float* WOb = (const float*)d_in[11];

  char* p = (char*)d_ws;
  auto take = [&](size_t bytes) -> char* {
    char* r = p;
    p += (bytes + 255) & ~(size_t)255;
    return r;
  };
  __bf16* Kin  = (__bf16*)take((size_t)BT_ * DM_ * 2);          // 4 MB
  __bf16* Qin  = (__bf16*)take((size_t)BT_ * DM_ * 2);          // 4 MB
  __bf16* Vin  = (__bf16*)take((size_t)BT_ * DM_ * 2);          // 4 MB
  __bf16* WK16 = (__bf16*)take((size_t)HDK_ * DM_ * 2);         // 128 KB
  __bf16* WQ16 = (__bf16*)take((size_t)HDK_ * DM_ * 2);         // 128 KB
  __bf16* WV16 = (__bf16*)take((size_t)HDM_ * DM_ * 2);         // 1 MB
  __bf16* WO16 = (__bf16*)take((size_t)DM_ * HDM_ * 2);         // 1 MB
  __bf16* Kp   = (__bf16*)take((size_t)BT_ * HDK_ * 2);         // 4 MB
  __bf16* Qp   = (__bf16*)take((size_t)BT_ * HDK_ * 2);         // 4 MB
  __bf16* Vt   = (__bf16*)take((size_t)B_ * H_ * DM_ * T_ * 2); // 32 MB
  __bf16* Emb  = (__bf16*)take((size_t)BT_ * HDM_ * 2);         // 32 MB

  auto cvt = [&](const float* s, __bf16* d, int n) {
    int n4 = n >> 2;
    cvt_f32_bf16<<<dim3((n4 + 255) / 256), dim3(256), 0, stream>>>(s, d, n4);
  };
  cvt(key_in, Kin, BT_ * DM_);
  cvt(qry_in, Qin, BT_ * DM_);
  cvt(val_in, Vin, BT_ * DM_);
  cvt(WKw, WK16, HDK_ * DM_);
  cvt(WQw, WQ16, HDK_ * DM_);
  cvt(WVw, WV16, HDM_ * DM_);
  cvt(WOw, WO16, DM_ * HDM_);

  // K / Q projections: [8192,256] x [256,256]^T  -> bf16 row-major
  gemm_bias_kernel<0><<<dim3(BT_ / 128, HDK_ / 128), dim3(256), 0, stream>>>(
      Kin, WK16, WKb, (void*)Kp, HDK_, DM_);
  gemm_bias_kernel<0><<<dim3(BT_ / 128, HDK_ / 128), dim3(256), 0, stream>>>(
      Qin, WQ16, WQb, (void*)Qp, HDK_, DM_);
  // V projection: [8192,256] x [2048,256]^T -> transposed Vt[b][h][d][t]
  gemm_bias_kernel<2><<<dim3(BT_ / 128, HDM_ / 128), dim3(256), 0, stream>>>(
      Vin, WV16, WVb, (void*)Vt, HDM_, DM_);

  // fused attention: grid (q-tiles, heads, batch)
  flash_attn_kernel<<<dim3(T_ / 128, H_, B_), dim3(256), 0, stream>>>(
      Qp, Kp, Vt, Emb);

  // output projection: [8192,2048] x [256,2048]^T + b -> fp32 d_out
  gemm_bias_kernel<1><<<dim3(BT_ / 128, DM_ / 128), dim3(256), 0, stream>>>(
      Emb, WO16, WOb, d_out, DM_, HDM_);
}
